// SimpleTreeSAT_68504728371310
// MI455X (gfx1250) — matrix operand order repaired and checked by reading, not compiled, not run on hardware
//
#include <hip/hip_runtime.h>
#include <hip/hip_bf16.h>
#include <math.h>

// ---------------------------------------------------------------------------
// Types for CDNA5 WMMA (wave32, 16x16x32 bf16 -> f32)
// ---------------------------------------------------------------------------
typedef __bf16 bf16_t;
typedef __attribute__((ext_vector_type(16))) __bf16 v16bf;
typedef __attribute__((ext_vector_type(8)))  __bf16 v8bf;
typedef __attribute__((ext_vector_type(8)))  float  v8f;
typedef __attribute__((ext_vector_type(4)))  unsigned int v4u;
typedef __attribute__((ext_vector_type(8)))  int v8i;
typedef __attribute__((ext_vector_type(4)))  int v4i;

union FragU {
    v16bf full;
    v8bf  h8[2];
};

#define EMB 128
#define NOPS 256
#define NVARS 128
#define NN 384            // NOPS + NVARS
#define BS 32
#define ROWS (BS * NN)    // 12288
#define K1P 160           // E+4 padded to 5*32
#define H1 512            // concat hidden (2 * 256)
#define NG 512            // LSTM gates (4 * E)

// ---------------------------------------------------------------------------
// Fragment loaders (CDNA5 ISA 7.12.2 layouts, wave32)
// ---------------------------------------------------------------------------
__device__ __forceinline__ v16bf load_frag_a(const bf16_t* base, int lda) {
    const int lane = threadIdx.x & 31;
    const int r    = lane & 15;
    const int koff = (lane < 16) ? 0 : 8;
    const bf16_t* p = base + r * lda + koff;
    FragU f;
    f.h8[0] = *(const v8bf*)(p);
    f.h8[1] = *(const v8bf*)(p + 16);
    return f.full;
}

__device__ __forceinline__ v16bf load_frag_b(const bf16_t* base, int ldb) {
    const int lane = threadIdx.x & 31;
    const int n    = lane & 15;
    const int kb   = (lane < 16) ? 0 : 8;
    const bf16_t* p = base + n * ldb + kb;
    FragU f;
    f.h8[0] = *(const v8bf*)(p);
    f.h8[1] = *(const v8bf*)(p + 16);
    return f.full;
}

__device__ __forceinline__ v8f wmma_bf16(v16bf a, v16bf b, v8f c) {
    return __builtin_amdgcn_wmma_f32_16x16x32_bf16(
        false, a, false, b, (short)0, c, false, false);
}

__device__ __forceinline__ float sigmoidf_(float x) {
    return 1.0f / (1.0f + __expf(-x));
}

// ---------------------------------------------------------------------------
// Tensor Data Mover: stage a 2D bf16 tile (rows x width, row stride in
// elements) from global memory into LDS.  Builds D# group0/group1 per
// CDNA5 ISA (data_size=1 -> 2 bytes, type=2 "image",
// count=1).  Groups 2/3 zero (2D tensor).  Issue from ONE wave per block,
// then s_wait_tensorcnt + workgroup barrier before consuming.
// ---------------------------------------------------------------------------
__device__ __forceinline__ void tdm_load_2d(unsigned ldsAddr, const void* gptr,
                                            unsigned width, unsigned rows,
                                            unsigned rowStride) {
    unsigned long long ga = (unsigned long long)(size_t)gptr;
    v4u g0;
    g0[0] = 1u;                                            // count=1, user desc
    g0[1] = ldsAddr;                                       // lds_addr
    g0[2] = (unsigned)(ga & 0xFFFFFFFFull);                // global_addr[31:0]
    g0[3] = (unsigned)((ga >> 32) & 0x01FFFFFFull) | (2u << 30); // addr[56:32] | type=2
    v8i g1;
    g1[0] = (int)(1u << 16);                               // wg_mask=0, data_size=1 (2B)
    g1[1] = (int)(width << 16);                            // tensor_dim0[15:0]
    g1[2] = (int)((width >> 16) | (rows << 16));           // dim0 hi | tensor_dim1 lo
    g1[3] = (int)(width << 16);                            // tile_dim0 = width
    g1[4] = (int)rows;                                     // tile_dim1 = rows
    g1[5] = (int)rowStride;                                // tensor_dim0_stride lo
    g1[6] = 0;
    g1[7] = 0;
    v4i gz;  gz[0] = 0; gz[1] = 0; gz[2] = 0; gz[3] = 0;
#if __clang_major__ >= 23
    v8i gz8; gz8[0] = 0; gz8[1] = 0; gz8[2] = 0; gz8[3] = 0;
    gz8[4] = 0; gz8[5] = 0; gz8[6] = 0; gz8[7] = 0;
    __builtin_amdgcn_tensor_load_to_lds(g0, g1, gz, gz, gz8, 0);
#else
    __builtin_amdgcn_tensor_load_to_lds(g0, g1, gz, gz, 0);
#endif
}

// ---------------------------------------------------------------------------
// Prep kernels
// ---------------------------------------------------------------------------
__global__ void k_convert_pad(const float* __restrict__ src, bf16_t* __restrict__ dst,
                              int R, int C, int Cp) {
    int idx = blockIdx.x * 256 + threadIdx.x;
    if (idx >= R * Cp) return;
    int r = idx / Cp, c = idx - r * Cp;
    float v = (c < C) ? src[r * C + c] : 0.0f;
    dst[idx] = (bf16_t)v;
}

__global__ void k_build_adj(const float* __restrict__ mxs,
                            bf16_t* __restrict__ A, bf16_t* __restrict__ AT) {
    long long idx = (long long)blockIdx.x * 256 + threadIdx.x;
    if (idx >= (long long)BS * NN * NN) return;
    int k = (int)(idx % NN);
    int m = (int)((idx / NN) % NN);
    int b = (int)(idx / ((long long)NN * NN));
    float v  = (m < NOPS) ? mxs[((long long)b * NOPS + m) * NN + k] : 0.0f;
    float vt = (k < NOPS) ? mxs[((long long)b * NOPS + k) * NN + m] : 0.0f;
    A[idx]  = (bf16_t)v;
    AT[idx] = (bf16_t)vt;
}

__global__ void k_init_x0(bf16_t* __restrict__ X0,
                          const float* __restrict__ cons, const float* __restrict__ negs) {
    int idx = blockIdx.x * 256 + threadIdx.x;
    if (idx >= ROWS * K1P) return;
    int row = idx / K1P, c = idx - row * K1P;
    float v = 0.0f;
    if (c >= EMB && c < EMB + 4) {
        int b = row / NN, m = row - b * NN;
        int ty = c - EMB;
        if (m < NOPS) {
            float cv = cons[b * NOPS + m];
            float nv = negs[b * NOPS + m];
            v = (ty == 0) ? cv : (ty == 1) ? (1.0f - cv - nv) : (ty == 2) ? nv : 0.0f;
        } else {
            v = (ty == 3) ? 1.0f : 0.0f;
        }
    }
    X0[idx] = (bf16_t)v;
}

__global__ void k_zero_c(float* __restrict__ cbuf) {
    int idx = blockIdx.x * 256 + threadIdx.x;
    if (idx < ROWS * EMB) cbuf[idx] = 0.0f;
}

// ---------------------------------------------------------------------------
// Step kernel 1: message MLPs.  X0 tile staged to LDS by TDM; layer-1 A
// fragments come from LDS (shared by all 8 waves).
// ---------------------------------------------------------------------------
__global__ __launch_bounds__(256) void msg_kernel(
    const bf16_t* __restrict__ X0,    // [ROWS][K1P]
    const bf16_t* __restrict__ Wc1,   // [512][K1P]
    const float*  __restrict__ b11, const float* __restrict__ b21,
    const bf16_t* __restrict__ W12, const bf16_t* __restrict__ W22,
    const float*  __restrict__ b12, const float* __restrict__ b22,
    bf16_t* __restrict__ msg1T, bf16_t* __restrict__ msg2T)
{
    __shared__ bf16_t xt[16][K1P];
    __shared__ bf16_t hid[16][H1];

    const int g0   = blockIdx.x * 16;
    const int wave = threadIdx.x >> 5;
    const int lane = threadIdx.x & 31;
    const int ln   = lane & 15;
    const int mb   = (lane < 16) ? 0 : 8;

    if (wave == 0) {
        tdm_load_2d((unsigned)(size_t)&xt[0][0],
                    X0 + (long long)g0 * K1P, K1P, 16, K1P);
        __builtin_amdgcn_s_wait_tensorcnt(0);
    }
    __syncthreads();

    // ---- layer 1: 4 tiles of 16 cols per wave, K = 160 ----
    v8f acc[4] = {};
    for (int ks = 0; ks < 5; ++ks) {
        const int k0 = ks * 32;
        v16bf af = load_frag_a(&xt[0][0] + k0, K1P);
#pragma unroll
        for (int t = 0; t < 4; ++t) {
            const int n0 = wave * 64 + t * 16;
            v16bf bfrag = load_frag_b(Wc1 + (long long)n0 * K1P + k0, K1P);
            acc[t] = wmma_bf16(af, bfrag, acc[t]);
        }
    }
#pragma unroll
    for (int t = 0; t < 4; ++t) {
        const int n = wave * 64 + t * 16 + ln;
        const float bias = (n < 256) ? b11[n] : b21[n - 256];
#pragma unroll
        for (int r = 0; r < 8; ++r) {
            float v = acc[t][r] + bias;
            hid[mb + r][n] = (bf16_t)fmaxf(v, 0.0f);
        }
    }
    __syncthreads();

    // ---- layer 2: waves 0-3 -> msg1, waves 4-7 -> msg2; 2 tiles each ----
    const int half = wave >> 2;
    const int w2   = wave & 3;
    const bf16_t* W2  = half ? W22 : W12;
    const float*  b2  = half ? b22 : b12;
    bf16_t* dstT      = half ? msg2T : msg1T;

    v8f acc2[2] = {};
    for (int ks = 0; ks < 8; ++ks) {
        const int k0 = ks * 32;
        v16bf af = load_frag_a(&hid[0][0] + half * 256 + k0, H1);
#pragma unroll
        for (int t = 0; t < 2; ++t) {
            const int n0 = w2 * 32 + t * 16;
            v16bf bfrag = load_frag_b(W2 + n0 * 256 + k0, 256);
            acc2[t] = wmma_bf16(af, bfrag, acc2[t]);
        }
    }

    const int bb = g0 / NN;
    const int m0 = g0 - bb * NN;
#pragma unroll
    for (int t = 0; t < 2; ++t) {
        const int n = w2 * 32 + t * 16 + ln;
        const float bias = b2[n];
        FragU pack;
#pragma unroll
        for (int r = 0; r < 8; ++r) pack.h8[0][r] = (bf16_t)(acc2[t][r] + bias);
        *(v8bf*)&dstT[((long long)bb * EMB + n) * NN + m0 + mb] = pack.h8[0];
    }
}

// ---------------------------------------------------------------------------
// Step kernel 2: inp = mxs2 @ msg1 + mxs2^T @ msg2.  Both 16x384 adjacency
// tiles staged to LDS by TDM (one DMA each), shared by all 8 waves.
// ---------------------------------------------------------------------------
__global__ __launch_bounds__(256) void adj_kernel(
    const bf16_t* __restrict__ mxsA,  // [BS][NN][NN]
    const bf16_t* __restrict__ mxsAT, // [BS][NN][NN]
    const bf16_t* __restrict__ msg1T, // [BS][128][NN]
    const bf16_t* __restrict__ msg2T, // [BS][128][NN]
    bf16_t* __restrict__ inpB)        // [ROWS][128]
{
    __shared__ bf16_t at[2][16][NN];  // 24 KB

    const int g0   = blockIdx.x * 16;
    const int bb   = g0 / NN;
    const int m0   = g0 - bb * NN;
    const int wave = threadIdx.x >> 5;
    const int lane = threadIdx.x & 31;
    const int ln   = lane & 15;
    const int mb   = (lane < 16) ? 0 : 8;
    const int n0   = wave * 16;

    if (wave == 0) {
        tdm_load_2d((unsigned)(size_t)&at[0][0][0],
                    mxsA + ((long long)bb * NN + m0) * NN, NN, 16, NN);
        tdm_load_2d((unsigned)(size_t)&at[1][0][0],
                    mxsAT + ((long long)bb * NN + m0) * NN, NN, 16, NN);
        __builtin_amdgcn_s_wait_tensorcnt(0);
    }
    __syncthreads();

    v8f acc = {};
    const bf16_t* B1 = msg1T + ((long long)bb * EMB + n0) * NN;
    for (int ks = 0; ks < 12; ++ks) {
        const int k0 = ks * 32;
        v16bf af = load_frag_a(&at[0][0][0] + k0, NN);
        v16bf bfrag = load_frag_b(B1 + k0, NN);
        acc = wmma_bf16(af, bfrag, acc);
    }
    const bf16_t* B2 = msg2T + ((long long)bb * EMB + n0) * NN;
    for (int ks = 0; ks < 12; ++ks) {
        const int k0 = ks * 32;
        v16bf af = load_frag_a(&at[1][0][0] + k0, NN);
        v16bf bfrag = load_frag_b(B2 + k0, NN);
        acc = wmma_bf16(af, bfrag, acc);
    }

    const int n = n0 + ln;
#pragma unroll
    for (int r = 0; r < 8; ++r)
        inpB[(long long)(g0 + mb + r) * EMB + n] = (bf16_t)acc[r];
}

// ---------------------------------------------------------------------------
// Step kernel 3: gates GEMMs + fused LSTM elementwise.  inp and h tiles
// staged to LDS by TDM; gates staged in LDS for the elementwise phase.
// ---------------------------------------------------------------------------
__global__ __launch_bounds__(256) void lstm_kernel(
    const bf16_t* __restrict__ inpB,  // [ROWS][128]
    bf16_t* __restrict__ X0,          // [ROWS][K1P]; cols 0..127 = h (updated)
    float*  __restrict__ cbuf,        // [ROWS][128]
    const bf16_t* __restrict__ Wih,   // [512][128]
    const bf16_t* __restrict__ Whh,   // [512][128]
    const float* __restrict__ bih, const float* __restrict__ bhh)
{
    __shared__ bf16_t xin[16][EMB];   // inp tile  (4 KB)
    __shared__ bf16_t xh[16][K1P];    // h tile    (5 KB)
    __shared__ float  gl[16][NG];     // gates     (32 KB)

    const int g0   = blockIdx.x * 16;
    const int wave = threadIdx.x >> 5;
    const int lane = threadIdx.x & 31;
    const int ln   = lane & 15;
    const int mb   = (lane < 16) ? 0 : 8;

    if (wave == 0) {
        tdm_load_2d((unsigned)(size_t)&xin[0][0],
                    inpB + (long long)g0 * EMB, EMB, 16, EMB);
        tdm_load_2d((unsigned)(size_t)&xh[0][0],
                    X0 + (long long)g0 * K1P, K1P, 16, K1P);
        __builtin_amdgcn_s_wait_tensorcnt(0);
    }
    __syncthreads();

    v8f acc[4] = {};
    for (int ks = 0; ks < 4; ++ks) {
        const int k0 = ks * 32;
        v16bf af = load_frag_a(&xin[0][0] + k0, EMB);
#pragma unroll
        for (int t = 0; t < 4; ++t) {
            const int n0 = wave * 64 + t * 16;
            v16bf bfrag = load_frag_b(Wih + n0 * EMB + k0, EMB);
            acc[t] = wmma_bf16(af, bfrag, acc[t]);
        }
    }
    for (int ks = 0; ks < 4; ++ks) {
        const int k0 = ks * 32;
        v16bf af = load_frag_a(&xh[0][0] + k0, K1P);
#pragma unroll
        for (int t = 0; t < 4; ++t) {
            const int n0 = wave * 64 + t * 16;
            v16bf bfrag = load_frag_b(Whh + n0 * EMB + k0, EMB);
            acc[t] = wmma_bf16(af, bfrag, acc[t]);
        }
    }
#pragma unroll
    for (int t = 0; t < 4; ++t) {
        const int n = wave * 64 + t * 16 + ln;
        const float bias = bih[n] + bhh[n];
#pragma unroll
        for (int r = 0; r < 8; ++r) gl[mb + r][n] = acc[t][r] + bias;
    }
    __syncthreads();

    for (int i = threadIdx.x; i < 16 * EMB; i += 256) {
        const int m = i >> 7;
        const int e = i & 127;
        const float iG = gl[m][e];
        const float fG = gl[m][128 + e];
        const float gG = gl[m][256 + e];
        const float oG = gl[m][384 + e];
        const long long rc = (long long)(g0 + m) * EMB + e;
        const float cOld = cbuf[rc];
        const float cNew = sigmoidf_(fG) * cOld + sigmoidf_(iG) * tanhf(gG);
        const float hNew = sigmoidf_(oG) * tanhf(cNew);
        cbuf[rc] = cNew;
        X0[(long long)(g0 + m) * K1P + e] = (bf16_t)hNew;
    }
}

// ---------------------------------------------------------------------------
// Classifier
// ---------------------------------------------------------------------------
__global__ __launch_bounds__(128) void clf_kernel(
    const bf16_t* __restrict__ X0,
    const float* __restrict__ w1, const float* __restrict__ b1,
    const float* __restrict__ w2, float* __restrict__ out)
{
    __shared__ float s[128];
    const int b = blockIdx.x;
    const int j = threadIdx.x;
    const bf16_t* h = X0 + (long long)(b * NN) * K1P;
    float acc = b1[j];
    for (int k = 0; k < EMB; ++k) acc += (float)h[k] * w1[j * EMB + k];
    s[j] = fmaxf(acc, 0.0f) * w2[j];
    __syncthreads();
    for (int off = 64; off > 0; off >>= 1) {
        if (j < off) s[j] += s[j + off];
        __syncthreads();
    }
    if (j == 0) out[b] = sigmoidf_(s[0]);
}

// ---------------------------------------------------------------------------
// Host orchestration
// ---------------------------------------------------------------------------
extern "C" void kernel_launch(void* const* d_in, const int* in_sizes, int n_in,
                              void* d_out, int out_size, void* d_ws, size_t ws_size,
                              hipStream_t stream) {
    (void)in_sizes; (void)n_in; (void)out_size; (void)ws_size;

    const float* mxs      = (const float*)d_in[0];
    const float* cons     = (const float*)d_in[1];
    const float* negs     = (const float*)d_in[2];
    const float* msg1_1_w = (const float*)d_in[3];
    const float* msg1_1_b = (const float*)d_in[4];
    const float* msg1_2_w = (const float*)d_in[5];
    const float* msg1_2_b = (const float*)d_in[6];
    const float* msg2_1_w = (const float*)d_in[7];
    const float* msg2_1_b = (const float*)d_in[8];
    const float* msg2_2_w = (const float*)d_in[9];
    const float* msg2_2_b = (const float*)d_in[10];
    const float* lstm_w_ih = (const float*)d_in[11];
    const float* lstm_w_hh = (const float*)d_in[12];
    const float* lstm_b_ih = (const float*)d_in[13];
    const float* lstm_b_hh = (const float*)d_in[14];
    const float* clf1_w   = (const float*)d_in[15];
    const float* clf1_b   = (const float*)d_in[16];
    const float* clf2_w   = (const float*)d_in[17];

    char* ws = (char*)d_ws;
    size_t off = 0;
    auto alloc = [&](size_t bytes) -> void* {
        void* p = ws + off;
        off = (off + bytes + 255) & ~(size_t)255;
        return p;
    };

    bf16_t* Wc1   = (bf16_t*)alloc((size_t)H1 * K1P * 2);
    bf16_t* W12   = (bf16_t*)alloc((size_t)EMB * 256 * 2);
    bf16_t* W22   = (bf16_t*)alloc((size_t)EMB * 256 * 2);
    bf16_t* Wih   = (bf16_t*)alloc((size_t)NG * EMB * 2);
    bf16_t* Whh   = (bf16_t*)alloc((size_t)NG * EMB * 2);
    bf16_t* X0    = (bf16_t*)alloc((size_t)ROWS * K1P * 2);
    float*  cbuf  = (float*) alloc((size_t)ROWS * EMB * 4);
    bf16_t* msg1T = (bf16_t*)alloc((size_t)BS * EMB * NN * 2);
    bf16_t* msg2T = (bf16_t*)alloc((size_t)BS * EMB * NN * 2);
    bf16_t* inpB  = (bf16_t*)alloc((size_t)ROWS * EMB * 2);
    bf16_t* mxsA  = (bf16_t*)alloc((size_t)BS * NN * NN * 2);
    bf16_t* mxsAT = (bf16_t*)alloc((size_t)BS * NN * NN * 2);

    k_convert_pad<<<(256 * K1P + 255) / 256, 256, 0, stream>>>(msg1_1_w, Wc1, 256, 132, K1P);
    k_convert_pad<<<(256 * K1P + 255) / 256, 256, 0, stream>>>(msg2_1_w, Wc1 + 256 * K1P, 256, 132, K1P);
    k_convert_pad<<<(EMB * 256 + 255) / 256, 256, 0, stream>>>(msg1_2_w, W12, EMB, 256, 256);
    k_convert_pad<<<(EMB * 256 + 255) / 256, 256, 0, stream>>>(msg2_2_w, W22, EMB, 256, 256);
    k_convert_pad<<<(NG * EMB + 255) / 256, 256, 0, stream>>>(lstm_w_ih, Wih, NG, EMB, EMB);
    k_convert_pad<<<(NG * EMB + 255) / 256, 256, 0, stream>>>(lstm_w_hh, Whh, NG, EMB, EMB);

    {
        long long tot = (long long)BS * NN * NN;
        k_build_adj<<<(unsigned)((tot + 255) / 256), 256, 0, stream>>>(mxs, mxsA, mxsAT);
    }
    k_init_x0<<<(ROWS * K1P + 255) / 256, 256, 0, stream>>>(X0, cons, negs);
    k_zero_c<<<(ROWS * EMB + 255) / 256, 256, 0, stream>>>(cbuf);

    const int blocks = ROWS / 16;  // 768
    for (int s = 0; s < 16; ++s) {
        msg_kernel<<<blocks, 256, 0, stream>>>(X0, Wc1, msg1_1_b, msg2_1_b,
                                               W12, W22, msg1_2_b, msg2_2_b,
                                               msg1T, msg2T);
        adj_kernel<<<blocks, 256, 0, stream>>>(mxsA, mxsAT, msg1T, msg2T, inpB);
        lstm_kernel<<<blocks, 256, 0, stream>>>(inpB, X0, cbuf, Wih, Whh,
                                                lstm_b_ih, lstm_b_hh);
    }

    clf_kernel<<<BS, 128, 0, stream>>>(X0, clf1_w, clf1_b, clf2_w, (float*)d_out);
}
